// GATConv_14422500180357
// MI455X (gfx1250) — matrix operand order repaired and checked by reading, not compiled, not run on hardware
//
#include <hip/hip_runtime.h>

#define BS 8
#define NN 2048
#define DINN 256
#define HH 128

typedef __attribute__((ext_vector_type(16))) _Float16 v16h;
typedef __attribute__((ext_vector_type(8)))  float    v8f;
typedef __attribute__((ext_vector_type(4)))  unsigned int u32x4;
typedef __attribute__((ext_vector_type(8)))  int      i32x8;
typedef __attribute__((ext_vector_type(4)))  int      i32x4;

union V16U { uint4 q[2]; v16h h; };

#if __has_builtin(__builtin_amdgcn_tensor_load_to_lds) && \
    __has_builtin(__builtin_amdgcn_s_wait_tensorcnt)
#define USE_TDM 1
#else
#define USE_TDM 0
#endif

__device__ __forceinline__ float fast_exp(float x) {
  // exp(x) = exp2(x * log2(e))
  return __builtin_amdgcn_exp2f(x * 1.4426950408889634f);
}

#if USE_TDM
typedef __attribute__((address_space(3))) char lds_char;
__device__ __forceinline__ unsigned lds_off(void* p) {
  return (unsigned)(unsigned long long)(lds_char*)p;
}

// TDM: copy a 128-row x 64B tile (rows stride NN*2 bytes) from global into
// LDS with 16B padding appended after every 64B row (LDS pitch 80B).
// D# per CDNA5 ISA ch.8: group0 = {count, lds_addr, global_addr, type},
// group1 = {flags/pads, tensor dims, tile dims, strides}; groups 2/3 unused
// for 2D tensors (zero-filled, 6-arg toolchain form).
__device__ __forceinline__ void tdm_load_tile(const _Float16* gsrc,
                                              unsigned lds_byte_off) {
  unsigned long long ga = (unsigned long long)(size_t)gsrc;
  u32x4 g0;
  g0[0] = 1u;                                   // count=1, user descriptor
  g0[1] = lds_byte_off;                         // lds_addr
  g0[2] = (unsigned)(ga & 0xffffffffu);         // global_addr[31:0]
  g0[3] = (unsigned)((ga >> 32) & 0x01ffffffu)  // global_addr[56:32]
          | (2u << 30);                         // type = 2 ("image")
  i32x8 g1;
  g1[0] = (int)((2u << 16)      // data_size   = 4 bytes
              | (1u << 20)      // pad_enable
              | (3u << 22)      // pad_interval: pad after 16 DWORDs (64B)
              | (3u << 25));    // pad_amount : 4 DWORDs (16B)
  g1[1] = (int)(1024u << 16);   // tensor_dim0 = 1024 DWORDs (low16)
  g1[2] = (int)(128u << 16);    // tensor_dim0 hi16=0 | tensor_dim1=128 lo16
  g1[3] = (int)(16u << 16);     // tensor_dim1 hi16=0 | tile_dim0=16 DWORDs
  g1[4] = (int)(128u);          // tile_dim1 = 128 rows, tile_dim2 = 0
  g1[5] = (int)(1024u);         // tensor_dim0_stride[31:0] = 1024 DWORDs
  g1[6] = 0;                    // stride0 hi | dim1_stride lo
  g1[7] = 0;
  i32x4 gz4 = {0, 0, 0, 0};
  i32x8 gz8 = {0, 0, 0, 0, 0, 0, 0, 0};
  __builtin_amdgcn_tensor_load_to_lds(g0, g1, gz4, gz4, gz8, 0);
}
#endif

// ---------------------------------------------------------------------------
// Kernel 1: h = x @ W^T, written transposed as f16: hT[b][ho][n]
// Wave computes 16 rows x 128 cols. W staged in LDS (f16) in 2 stages of 64
// rows, row pitch 264 halves (528B) -> conflict-free ds_load_b128.
// ---------------------------------------------------------------------------
__global__ __launch_bounds__(256) void k_gemm_h(const float* __restrict__ x,
                                                const float* __restrict__ W,
                                                _Float16* __restrict__ hT) {
  __shared__ _Float16 Wsm[64 * 264];

  const int b    = blockIdx.y;
  const int wave = threadIdx.x >> 5;
  const int lane = threadIdx.x & 31;
  const int li   = lane & 15;
  const int grp  = lane >> 4;
  const int n0   = blockIdx.x * 128 + wave * 16;
  const int row  = n0 + li;
  const float* xrow = x + ((size_t)b * NN + row) * DINN;

  v8f zero = {0.f, 0.f, 0.f, 0.f, 0.f, 0.f, 0.f, 0.f};
  v8f acc[8];
#pragma unroll
  for (int c = 0; c < 8; ++c) acc[c] = zero;

  for (int stage = 0; stage < 2; ++stage) {
    // Stage W rows [stage*64, stage*64+64) into LDS as f16.
    for (int it = threadIdx.x; it < 64 * 64; it += 256) {
      int r = it >> 6, qc = it & 63;
      float4 wq = *(const float4*)(W + (size_t)(stage * 64 + r) * DINN + qc * 4);
      _Float16* dst = &Wsm[r * 264 + qc * 4];
      dst[0] = (_Float16)wq.x; dst[1] = (_Float16)wq.y;
      dst[2] = (_Float16)wq.z; dst[3] = (_Float16)wq.w;
    }
    __syncthreads();

#pragma unroll
    for (int kt = 0; kt < 8; ++kt) {
      // A operand: rows of x, K = kt*32 + 8*grp + {0..7, 16..23}
      const float* ab = xrow + kt * 32 + grp * 8;
      float4 q0 = *(const float4*)(ab);
      float4 q1 = *(const float4*)(ab + 4);
      float4 q2 = *(const float4*)(ab + 16);
      float4 q3 = *(const float4*)(ab + 20);
      v16h av;
      av[0]  = (_Float16)q0.x; av[1]  = (_Float16)q0.y;
      av[2]  = (_Float16)q0.z; av[3]  = (_Float16)q0.w;
      av[4]  = (_Float16)q1.x; av[5]  = (_Float16)q1.y;
      av[6]  = (_Float16)q1.z; av[7]  = (_Float16)q1.w;
      av[8]  = (_Float16)q2.x; av[9]  = (_Float16)q2.y;
      av[10] = (_Float16)q2.z; av[11] = (_Float16)q2.w;
      av[12] = (_Float16)q3.x; av[13] = (_Float16)q3.y;
      av[14] = (_Float16)q3.z; av[15] = (_Float16)q3.w;

#pragma unroll
      for (int c = 0; c < 4; ++c) {
        // B operand: B[d][ko] = W[ko][d]; lane holds col ko = 16c+li,
        // K = kt*32 + 8*grp + {0..7, 16..23} in LDS row.
        const _Float16* bp = &Wsm[(c * 16 + li) * 264 + kt * 32 + grp * 8];
        V16U bu;
        bu.q[0] = *(const uint4*)(bp);
        bu.q[1] = *(const uint4*)(bp + 16);
        acc[stage * 4 + c] = __builtin_amdgcn_wmma_f32_16x16x32_f16(
            false, av, false, bu.h, (short)0, acc[stage * 4 + c], false, false);
      }
    }
    __syncthreads();
  }

  // Write transposed f16: hT[b][ko][n]. C layout: lane l, reg r ->
  // row n0 + r + 8*grp, col ko (chunk) + li.
#pragma unroll
  for (int ch = 0; ch < 8; ++ch) {
    int ko = (ch >> 2) * 64 + (ch & 3) * 16 + li;
#pragma unroll
    for (int r = 0; r < 8; ++r) {
      int n = n0 + r + grp * 8;
      hT[((size_t)b * HH + ko) * NN + n] = (_Float16)acc[ch][r];
    }
  }
}

// ---------------------------------------------------------------------------
// Kernel 2: s_self[b][n] = h[b][n] . a[0:128], s_nb = h . a[128:256]
// Coalesced along n (hT is [b][ho][n]).
// ---------------------------------------------------------------------------
__global__ __launch_bounds__(256) void k_scores(const _Float16* __restrict__ hT,
                                                const float* __restrict__ a,
                                                float* __restrict__ s_self,
                                                float* __restrict__ s_nb) {
  const int b = blockIdx.y;
  const int n = blockIdx.x * 256 + threadIdx.x;
  float ss = 0.f, sn = 0.f;
#pragma unroll 4
  for (int ho = 0; ho < HH; ++ho) {
    float v = (float)hT[((size_t)b * HH + ho) * NN + n];
    ss += v * a[ho];
    sn += v * a[HH + ho];
  }
  s_self[b * NN + n] = ss;
  s_nb[b * NN + n]   = sn;
}

// ---------------------------------------------------------------------------
// Kernel 3: fused masked-softmax attention + P@V + ELU.
// Single-pass softmax: m_i = LeakyReLU(s_self_i + max_j s_nb_j) is a per-row
// upper bound (LeakyReLU monotonic), so no online rescaling is needed.
// V tile (32x128 f16, stored transposed with 80B pitch) is staged by the
// Tensor Data Mover, double-buffered, with s_wait_tensorcnt for completion.
// ---------------------------------------------------------------------------
__global__ __launch_bounds__(256) void k_attn(const float* __restrict__ adj,
                                              const _Float16* __restrict__ hT,
                                              const float* __restrict__ s_self,
                                              const float* __restrict__ s_nb,
                                              float* __restrict__ out) {
  __shared__ float    snb_sm[NN];          // 8 KB
  __shared__ _Float16 bt[2][128 * 40];     // 2 x 10 KB, pitch 40 halves (80B)
  __shared__ float    red[8];

  const int b    = blockIdx.y;
  const int wave = threadIdx.x >> 5;
  const int lane = threadIdx.x & 31;
  const int li   = lane & 15;
  const int grp  = lane >> 4;
  const int i0   = blockIdx.x * 128 + wave * 16;
  const int i    = i0 + li;

  const _Float16* hbase = hT + (size_t)b * HH * NN;   // [ho][n] rows, pitch NN

#if USE_TDM
  if (wave == 0) tdm_load_tile(hbase, lds_off(&bt[0][0]));   // prologue: tile 0
#endif

  // Stage s_nb row into LDS; find its max (for the softmax bound).
  float lmax = -3.4e38f;
  const float* snbg = s_nb + b * NN;
  for (int q = threadIdx.x; q < NN / 4; q += 256) {
    float4 v = *(const float4*)(snbg + q * 4);
    *(float4*)&snb_sm[q * 4] = v;
    lmax = fmaxf(lmax, fmaxf(fmaxf(v.x, v.y), fmaxf(v.z, v.w)));
  }
#pragma unroll
  for (int off = 16; off >= 1; off >>= 1)
    lmax = fmaxf(lmax, __shfl_xor(lmax, off, 32));
  if (lane == 0) red[wave] = lmax;
  __syncthreads();
  float cmax = red[0];
#pragma unroll
  for (int w = 1; w < 8; ++w) cmax = fmaxf(cmax, red[w]);

  const float ss = s_self[b * NN + i];
  const float t0 = ss + cmax;
  const float mi = fmaxf(t0, 0.2f * t0);   // LeakyReLU of the bound
  float lsum = 0.f;
  const float* adjrow = adj + ((size_t)b * NN + i) * NN;

  v8f zero = {0.f, 0.f, 0.f, 0.f, 0.f, 0.f, 0.f, 0.f};
  v8f acc[8];
#pragma unroll
  for (int c = 0; c < 8; ++c) acc[c] = zero;

  for (int j0 = 0; j0 < NN; j0 += 32) {
    const int buf = (j0 >> 5) & 1;
    __syncthreads();   // everyone done reading the buffer we are about to fill
#if USE_TDM
    if (wave == 0) {
      // Issue next tile into the other buffer (wrap-around dummy on the last
      // iteration keeps the tensorcnt<=1 wait exact for the current tile).
      int nj = (j0 + 32) & (NN - 1);
      tdm_load_tile(hbase + nj, lds_off(&bt[buf ^ 1][0]));
      __builtin_amdgcn_s_wait_tensorcnt(1);   // current tile has landed
    }
#else
    // Manual staging: bt[buf][ho][j_local] <- hT[b][ho][j0 + j_local]
#pragma unroll
    for (int p = 0; p < 2; ++p) {
      int idx  = p * 256 + threadIdx.x;   // 512 x 16B chunks
      int ho   = idx >> 2;
      int part = idx & 3;
      uint4 v = *(const uint4*)(hbase + (size_t)ho * NN + j0 + part * 8);
      *(uint4*)&bt[buf][ho * 40 + part * 8] = v;
    }
#endif
    __syncthreads();

    // Prefetch the next adj tile while we compute on this one.
    __builtin_prefetch(adjrow + j0 + 32, 0, 1);

    // P tile (A operand): lane covers row i, j = j0 + 8*grp + {0..7, 16..23}
    const float* ap = adjrow + j0 + grp * 8;
    const float* sp = &snb_sm[j0 + grp * 8];
    float4 a0 = *(const float4*)(ap);
    float4 a1 = *(const float4*)(ap + 4);
    float4 a2 = *(const float4*)(ap + 16);
    float4 a3 = *(const float4*)(ap + 20);
    float4 s0 = *(const float4*)(sp);
    float4 s1 = *(const float4*)(sp + 4);
    float4 s2 = *(const float4*)(sp + 16);
    float4 s3 = *(const float4*)(sp + 20);

    auto pfun = [&](float adjv, float snv) -> _Float16 {
      float e = ss + snv;
      e = fmaxf(e, 0.2f * e);                      // LeakyReLU (alpha<1)
      float pv = (adjv == 0.f) ? 0.f : fast_exp(e - mi);
      lsum += pv;
      return (_Float16)pv;
    };
    v16h av;
    av[0]  = pfun(a0.x, s0.x); av[1]  = pfun(a0.y, s0.y);
    av[2]  = pfun(a0.z, s0.z); av[3]  = pfun(a0.w, s0.w);
    av[4]  = pfun(a1.x, s1.x); av[5]  = pfun(a1.y, s1.y);
    av[6]  = pfun(a1.z, s1.z); av[7]  = pfun(a1.w, s1.w);
    av[8]  = pfun(a2.x, s2.x); av[9]  = pfun(a2.y, s2.y);
    av[10] = pfun(a2.z, s2.z); av[11] = pfun(a2.w, s2.w);
    av[12] = pfun(a3.x, s3.x); av[13] = pfun(a3.y, s3.y);
    av[14] = pfun(a3.z, s3.z); av[15] = pfun(a3.w, s3.w);

#pragma unroll
    for (int c = 0; c < 8; ++c) {
      // B operand: bt[buf][ho = 16c+li][j_local = 8*grp + {0..7, 16..23}]
      const _Float16* bp = &bt[buf][(c * 16 + li) * 40 + grp * 8];
      V16U bu;
      bu.q[0] = *(const uint4*)(bp);
      bu.q[1] = *(const uint4*)(bp + 16);
      acc[c] = __builtin_amdgcn_wmma_f32_16x16x32_f16(
          false, av, false, bu.h, (short)0, acc[c], false, false);
    }
  }

  // Row sums: lane l and l^16 each hold half of row (l&15).
  lsum += __shfl_xor(lsum, 16, 32);

#pragma unroll
  for (int r = 0; r < 8; ++r) {
    float inv = 1.0f / __shfl(lsum, r + grp * 8, 32);
    int orow = i0 + r + grp * 8;
    float* op = out + ((size_t)b * NN + orow) * HH;
#pragma unroll
    for (int c = 0; c < 8; ++c) {
      float o = acc[c][r] * inv;
      o = (o > 0.f) ? o : (fast_exp(o) - 1.0f);   // ELU (alpha=1)
      op[c * 16 + li] = o;
    }
  }
}

// ---------------------------------------------------------------------------
extern "C" void kernel_launch(void* const* d_in, const int* in_sizes, int n_in,
                              void* d_out, int out_size, void* d_ws, size_t ws_size,
                              hipStream_t stream) {
  (void)in_sizes; (void)n_in; (void)out_size; (void)ws_size;
  const float* x   = (const float*)d_in[0];
  const float* adj = (const float*)d_in[1];
  const float* W   = (const float*)d_in[2];
  const float* a   = (const float*)d_in[3];
  float* out = (float*)d_out;

  char* ws = (char*)d_ws;
  _Float16* hT     = (_Float16*)ws;                                 // 4 MB
  float*    s_self = (float*)(ws + (size_t)BS * HH * NN * sizeof(_Float16));
  float*    s_nb   = s_self + BS * NN;

  k_gemm_h<<<dim3(16, 8), 256, 0, stream>>>(x, W, hT);
  k_scores<<<dim3(8, 8), 256, 0, stream>>>(hT, a, s_self, s_nb);
  k_attn  <<<dim3(16, 8), 256, 0, stream>>>(adj, hT, s_self, s_nb, out);
}
